// DKVMN_87540023427714
// MI455X (gfx1250) — matrix operand order repaired and checked by reading, not compiled, not run on hardware
//
#include <hip/hip_runtime.h>
#include <hip/hip_bf16.h>
#include <math.h>

#define BDIM   128
#define LDIM   200
#define HDIM   128
#define CDIM   64
#define QNUM   10000

typedef __attribute__((ext_vector_type(16))) _Float16 v16h;
typedef __attribute__((ext_vector_type(8)))  _Float16 v8h;
typedef __attribute__((ext_vector_type(8)))  float    v8f;

// ---------------------------------------------------------------------------
// One-time f32 -> f16 pre-conversion (bandwidth-trivial: ~16 MB total).
// ---------------------------------------------------------------------------
__global__ __launch_bounds__(256) void cvt_f32_f16(
    const float* __restrict__ src, _Float16* __restrict__ dst, int n)
{
    int i = blockIdx.x * 256 + threadIdx.x;
    if (i < n) dst[i] = (_Float16)src[i];
}

// W1q = p_W1[:, :H]  (strip out of the (H, 2H) row-major matrix)
__global__ __launch_bounds__(256) void cvt_w1q_f16(
    const float* __restrict__ pW1, _Float16* __restrict__ dst)
{
    int i = blockIdx.x * 256 + threadIdx.x;          // 0 .. 16383
    int n = i >> 7, k = i & 127;
    dst[i] = (_Float16)pW1[n * (2 * HDIM) + k];
}

// ---------------------------------------------------------------------------
// 16x16 f32 tile = A(16x128 f16) @ W^T(128x16) via 4 chained WMMAs.
// Weights are pre-converted f16, row-major, ld = 128.  B operand for chunk kc
// holds K {8hi+0..7} and {8hi+16..23} (+32*kc) -> two aligned 16B loads.
// ---------------------------------------------------------------------------
__device__ __forceinline__ v8f gemm_tile_k128(const v16h a[4],
                                              const _Float16* __restrict__ wrow,
                                              int hi) {
    v8f acc = {0.f, 0.f, 0.f, 0.f, 0.f, 0.f, 0.f, 0.f};
#pragma unroll
    for (int kc = 0; kc < 4; ++kc) {
        v8h blo = *(const v8h*)(wrow + 32 * kc + 8 * hi);
        v8h bhs = *(const v8h*)(wrow + 32 * kc + 8 * hi + 16);
        v16h b = __builtin_shufflevector(blo, bhs, 0, 1, 2, 3, 4, 5, 6, 7,
                                         8, 9, 10, 11, 12, 13, 14, 15);
        acc = __builtin_amdgcn_wmma_f32_16x16x32_f16(
            /*neg_a=*/false, a[kc], /*neg_b=*/false, b,
            /*c_mod=*/(short)0, acc, /*reuse_a=*/false, /*reuse_b=*/false);
    }
    return acc;
}

// A-operand chunk: lane m = lane&15 holds K pairs {8hi+0..7} and {8hi+16..23}
// (+32*kc) -> two aligned 16B loads concatenated.
__device__ __forceinline__ v16h load_a_chunk(const _Float16* __restrict__ row,
                                             int kc, int hi) {
    const int kb = 32 * kc + 8 * hi;
    v8h lo = *(const v8h*)(row + kb);
    v8h hs = *(const v8h*)(row + kb + 16);
    return __builtin_shufflevector(lo, hs, 0, 1, 2, 3, 4, 5, 6, 7,
                                   8, 9, 10, 11, 12, 13, 14, 15);
}

__device__ __forceinline__ float sigmoidf_fast(float x) {
    return 1.f / (1.f + __expf(-x));
}

// ---------------------------------------------------------------------------
// Kernel 1: fused gather + 4 WMMA GEMMs + softmax/sigmoid/tanh epilogues.
// One wave per 16-row tile of the flattened (B*L) rows.
// ---------------------------------------------------------------------------
__global__ __launch_bounds__(32) void dkvmn_embed_wmma(
    const int* __restrict__ X, const int* __restrict__ Q,
    const _Float16* __restrict__ q16, const _Float16* __restrict__ x16,
    const _Float16* __restrict__ key16, const _Float16* __restrict__ w1q16,
    const _Float16* __restrict__ e16, const _Float16* __restrict__ a16,
    const float* __restrict__ p_b1, const float* __restrict__ e_b,
    const float* __restrict__ a_b,
    float* __restrict__ Kout, float* __restrict__ Qproj,
    float* __restrict__ Eout, float* __restrict__ Aout)
{
    __shared__ float logits[16][CDIM];

    const int lane = threadIdx.x;       // 0..31
    const int mym  = lane & 15;         // row (A) / col (B/C/D) within tile
    const int hi   = lane >> 4;         // half-wave select
    const int r0   = blockIdx.x * 16;   // first flattened row of this tile

    const int row = r0 + mym;
    const _Float16* __restrict__ qrow = q16 + (size_t)Q[row] * HDIM;
    const _Float16* __restrict__ xrow = x16 + (size_t)X[row] * HDIM;

    v16h aQ[4], aX[4];
#pragma unroll
    for (int kc = 0; kc < 4; ++kc) {
        aQ[kc] = load_a_chunk(qrow, kc, hi);
        aX[kc] = load_a_chunk(xrow, kc, hi);
    }

    // ---- K logits: Qe @ key_W^T  (16 x 64) ----
#pragma unroll
    for (int nt = 0; nt < 4; ++nt) {
        const int col = nt * 16 + mym;
        v8f acc = gemm_tile_k128(aQ, key16 + (size_t)col * HDIM, hi);
#pragma unroll
        for (int v = 0; v < 8; ++v)
            logits[v + hi * 8][col] = acc[v];
    }
    __syncthreads();

    // ---- softmax over C=64, one row per lane 0..15 ----
    if (lane < 16) {
        float mx = -3.0e38f;
        for (int c = 0; c < CDIM; ++c) mx = fmaxf(mx, logits[lane][c]);
        float s = 0.f;
        for (int c = 0; c < CDIM; ++c) s += __expf(logits[lane][c] - mx);
        const float inv = 1.f / s;
        float* krow = Kout + (size_t)(r0 + lane) * CDIM;
        for (int c = 0; c < CDIM; ++c)
            krow[c] = __expf(logits[lane][c] - mx) * inv;
    }
    __syncthreads();

    // ---- Qproj = Qe @ W1q^T + b1 ----
#pragma unroll
    for (int nt = 0; nt < 8; ++nt) {
        const int col = nt * 16 + mym;
        v8f acc = gemm_tile_k128(aQ, w1q16 + (size_t)col * HDIM, hi);
        const float bias = p_b1[col];
#pragma unroll
        for (int v = 0; v < 8; ++v)
            Qproj[(size_t)(r0 + v + hi * 8) * HDIM + col] = acc[v] + bias;
    }

    // ---- E = sigmoid(Xe @ e_W^T + e_b) ----
#pragma unroll
    for (int nt = 0; nt < 8; ++nt) {
        const int col = nt * 16 + mym;
        v8f acc = gemm_tile_k128(aX, e16 + (size_t)col * HDIM, hi);
        const float bias = e_b[col];
#pragma unroll
        for (int v = 0; v < 8; ++v)
            Eout[(size_t)(r0 + v + hi * 8) * HDIM + col] =
                sigmoidf_fast(acc[v] + bias);
    }

    // ---- A = tanh(Xe @ a_W^T + a_b) ----
#pragma unroll
    for (int nt = 0; nt < 8; ++nt) {
        const int col = nt * 16 + mym;
        v8f acc = gemm_tile_k128(aX, a16 + (size_t)col * HDIM, hi);
        const float bias = a_b[col];
#pragma unroll
        for (int v = 0; v < 8; ++v)
            Aout[(size_t)(r0 + v + hi * 8) * HDIM + col] =
                tanhf(acc[v] + bias);
    }
}

// ---------------------------------------------------------------------------
// Kernel 2: sequential scan. One workgroup per batch element; memory state
// M (64x128 f32 = 32 KB) lives in LDS for all 200 steps. All streaming
// operands (K/Qproj/E/A, 46 MB total) are L2-resident on MI455X (192 MB L2).
// ---------------------------------------------------------------------------
__global__ __launch_bounds__(HDIM) void dkvmn_scan(
    const float* __restrict__ Kbuf, const float* __restrict__ Qproj,
    const float* __restrict__ Ebuf, const float* __restrict__ Abuf,
    const float* __restrict__ p_W1, const float* __restrict__ p_W2,
    const float* __restrict__ p_b2, float* __restrict__ out)
{
    __shared__ float M[CDIM][HDIM];   // 32 KB memory state
    __shared__ float Kt[CDIM];
    __shared__ float Rt[HDIM];
    __shared__ float red[HDIM];

    const int h = threadIdx.x;        // 0..127, owns feature h
    const int b = blockIdx.x;         // batch element

    for (int c = 0; c < CDIM; ++c) M[c][h] = 0.f;

    const float  w2 = p_W2[h];
    const float  b2 = p_b2[0];
    const float* __restrict__ w1r = p_W1 + (size_t)h * (2 * HDIM) + HDIM;
    __syncthreads();

    for (int t = 0; t < LDIM; ++t) {
        const size_t r = (size_t)b * LDIM + t;
        if (h < CDIM) Kt[h] = Kbuf[r * CDIM + h];
        const float Et = Ebuf[r * HDIM + h];
        const float At = Abuf[r * HDIM + h];
        const float Qp = Qproj[r * HDIM + h];
        __syncthreads();

        // Rt = K_t . M  (read vector)
        float rv = 0.f;
#pragma unroll 8
        for (int c = 0; c < CDIM; ++c) rv += Kt[c] * M[c][h];
        Rt[h] = rv;
        __syncthreads();

        // h_vec = tanh(Qproj + Rt @ W1r^T), then dot with p_W2
        float s = Qp;
#pragma unroll 8
        for (int k = 0; k < HDIM; ++k) s += Rt[k] * w1r[k];
        red[h] = tanhf(s) * w2;
        __syncthreads();

        for (int off = HDIM / 2; off > 0; off >>= 1) {
            if (h < off) red[h] += red[h + off];
            __syncthreads();
        }
        if (h == 0)
            out[(size_t)b * LDIM + t] = sigmoidf_fast(red[0] + b2);

        // M <- M * (1 - K (x) E) + K (x) A
#pragma unroll 8
        for (int c = 0; c < CDIM; ++c) {
            const float kc = Kt[c];
            M[c][h] = M[c][h] * (1.f - kc * Et) + kc * At;
        }
        __syncthreads();
    }
}

// ---------------------------------------------------------------------------
extern "C" void kernel_launch(void* const* d_in, const int* in_sizes, int n_in,
                              void* d_out, int out_size, void* d_ws, size_t ws_size,
                              hipStream_t stream) {
    (void)in_sizes; (void)n_in; (void)out_size; (void)ws_size;

    const int*   X     = (const int*)  d_in[0];
    const int*   Q     = (const int*)  d_in[1];
    const float* q_emb = (const float*)d_in[2];
    const float* x_emb = (const float*)d_in[3];
    const float* key_W = (const float*)d_in[4];
    const float* p_W1  = (const float*)d_in[5];
    const float* p_b1  = (const float*)d_in[6];
    const float* p_W2  = (const float*)d_in[7];
    const float* p_b2  = (const float*)d_in[8];
    const float* e_W   = (const float*)d_in[9];
    const float* e_b   = (const float*)d_in[10];
    const float* a_W   = (const float*)d_in[11];
    const float* a_b   = (const float*)d_in[12];

    // -------- workspace layout --------
    // f32:  K | Qproj | E | A                     (~45.9 MB)
    // f16:  q_emb | x_emb | key_W | W1q | e_W | a_W  (~7.8 MB)
    float* ws    = (float*)d_ws;
    float* Kbuf  = ws;
    float* Qproj = Kbuf  + (size_t)BDIM * LDIM * CDIM;
    float* Ebuf  = Qproj + (size_t)BDIM * LDIM * HDIM;
    float* Abuf  = Ebuf  + (size_t)BDIM * LDIM * HDIM;

    _Float16* q16   = (_Float16*)(Abuf + (size_t)BDIM * LDIM * HDIM);
    _Float16* x16   = q16   + (size_t)QNUM * HDIM;          // 1,280,000
    _Float16* key16 = x16   + (size_t)2 * QNUM * HDIM;      // 2,560,000
    _Float16* w1q16 = key16 + (size_t)CDIM * HDIM;          // 8,192
    _Float16* e16   = w1q16 + (size_t)HDIM * HDIM;          // 16,384
    _Float16* a16   = e16   + (size_t)HDIM * HDIM;          // 16,384

    // -------- one-time weight / embedding conversion --------
    const int nq = QNUM * HDIM, nx = 2 * QNUM * HDIM;
    const int nk = CDIM * HDIM, nh = HDIM * HDIM;
    cvt_f32_f16<<<(nq + 255) / 256, 256, 0, stream>>>(q_emb, q16, nq);
    cvt_f32_f16<<<(nx + 255) / 256, 256, 0, stream>>>(x_emb, x16, nx);
    cvt_f32_f16<<<(nk + 255) / 256, 256, 0, stream>>>(key_W, key16, nk);
    cvt_f32_f16<<<(nh + 255) / 256, 256, 0, stream>>>(e_W, e16, nh);
    cvt_f32_f16<<<(nh + 255) / 256, 256, 0, stream>>>(a_W, a16, nh);
    cvt_w1q_f16<<<(nh + 255) / 256, 256, 0, stream>>>(p_W1, w1q16);

    // -------- fused gather + WMMA GEMMs --------
    const int tiles = (BDIM * LDIM) / 16;    // 1600 wave-tiles
    dkvmn_embed_wmma<<<tiles, 32, 0, stream>>>(
        X, Q, q16, x16, key16, w1q16, e16, a16,
        p_b1, e_b, a_b, Kbuf, Qproj, Ebuf, Abuf);

    // -------- sequential scan --------
    dkvmn_scan<<<BDIM, HDIM, 0, stream>>>(
        Kbuf, Qproj, Ebuf, Abuf, p_W1, p_W2, p_b2, (float*)d_out);
}
